// FDConv_47691316854813
// MI455X (gfx1250) — matrix operand order repaired and checked by reading, compiled, not run on hardware
//
#include <hip/hip_runtime.h>

// ---------------------------------------------------------------------------
// FDConv on MI455X (gfx1250, wave32, WMMA).
//
// Roofline: grouped conv = 38.6 GFLOP over >=270 MB HBM traffic -> bandwidth
// bound (~11.6 us @ 23.3 TB/s). Strategy: single pass over x, implicit GEMM
// with v_wmma_f32_16x16x32_bf16 (bf16 mandatory: spatial kernels ~1e-9, below
// fp16 denormal range), all operands staged in LDS.
// Prologue (permutation + 2x separable inverse DFT) is < 1 GFLOP.
// ---------------------------------------------------------------------------

#define NFREQ 18624   // D1 * D2R
#define D1    192
#define D2    192
#define D2R   97
#define BB    8
#define CIN   64
#define COUT  64
#define HH    256
#define WW    256

typedef __attribute__((ext_vector_type(16))) __bf16 v16bf;
typedef __attribute__((ext_vector_type(8)))  float  v8f;

__device__ __forceinline__ unsigned short f2bf(float f) {
  // round-to-nearest-even f32 -> bf16
  unsigned u = __float_as_uint(f);
  u += 0x7FFFu + ((u >> 16) & 1u);
  return (unsigned short)(u >> 16);
}

// ---- 1) exact float64 radial distances (replicates np.fft.fftfreq/rfftfreq,
//         dist = sqrt(fh^2 + fw^2) with identical IEEE op order) -------------
__global__ void k_dist(double* __restrict__ dist) {
#pragma clang fp contract(off)
  int p = blockIdx.x * 256 + threadIdx.x;
  if (p >= NFREQ) return;
  int h = p / D2R, w = p % D2R;
  double val = 1.0 / 192.0;  // numpy: val = 1.0/(n*d), then integer*val
  double a = (double)((h < 96) ? h : h - 192) * val;
  double b = (double)w * val;
  double aa = a * a;
  double bb = b * b;
  dist[p] = sqrt(aa + bb);
}

// ---- 2) stable argsort rank: rank[p] = #{q : (dist_q, q) < (dist_p, p)} ----
//  dist >= 0, so the uint64 bit pattern orders identically to the value.
__global__ void k_rank(const double* __restrict__ dist, int* __restrict__ rank) {
  __shared__ unsigned long long sd[256];
  int p = blockIdx.x * 256 + threadIdx.x;
  unsigned long long my = 0;
  if (p < NFREQ) my = (unsigned long long)__double_as_longlong(dist[p]);
  int r = 0;
  for (int base = 0; base < NFREQ; base += 256) {
    int q = base + threadIdx.x;
    sd[threadIdx.x] = (q < NFREQ) ? (unsigned long long)__double_as_longlong(dist[q]) : ~0ull;
    __syncthreads();
    int lim = (NFREQ - base < 256) ? (NFREQ - base) : 256;
    for (int j = 0; j < lim; ++j) {
      unsigned long long dq = sd[j];
      r += (int)((dq < my) || (dq == my && (base + j) < p));
    }
    __syncthreads();
  }
  if (p < NFREQ) rank[p] = r;
}

// ---- 3) kernel-attention weighted coeffs, gathered into the freq map -------
//  fmap[b, p] = sum_k sigmoid(logit)*0.5 * dft_weight[k, rank[p], :]
__global__ void k_fmap(const float* __restrict__ logits, const float* __restrict__ dw,
                       const int* __restrict__ rank, float* __restrict__ fmap) {
  int idx = blockIdx.x * 256 + threadIdx.x;
  if (idx >= BB * NFREQ) return;
  int b = idx / NFREQ, p = idx % NFREQ;
  int n = rank[p];
  float re = 0.f, im = 0.f;
  for (int k = 0; k < 4; ++k) {
    float a = 0.5f / (1.f + expf(-logits[b * 4 + k]));  // sigmoid * 2/KNUM
    re += a * dw[(k * NFREQ + n) * 2 + 0];
    im += a * dw[(k * NFREQ + n) * 2 + 1];
  }
  fmap[idx * 2 + 0] = re;
  fmap[idx * 2 + 1] = im;
}

// ---- 4) inverse DFT along axis 0 (complex, length 192), incl 1/192 ---------
__global__ void k_idft_h(const float* __restrict__ fmap, float* __restrict__ G) {
  __shared__ float tc[D1], tsn[D1];
  if (threadIdx.x < D1) {
    float ang = (6.2831853071795864769f / 192.f) * (float)threadIdx.x;
    float s, c;
    sincosf(ang, &s, &c);
    tc[threadIdx.x] = c;
    tsn[threadIdx.x] = s;
  }
  __syncthreads();
  int idx = blockIdx.x * 256 + threadIdx.x;
  if (idx >= BB * D1 * D2R) return;
  int w = idx % D2R;
  int u = (idx / D2R) % D1;
  int b = idx / (D2R * D1);
  const float* f = fmap + (size_t)(b * NFREQ + w) * 2;
  float gr = 0.f, gi = 0.f;
  int t = 0;
  for (int h = 0; h < D1; ++h) {
    float fr = f[h * D2R * 2 + 0];
    float fi = f[h * D2R * 2 + 1];
    float c = tc[t], s = tsn[t];
    gr += fr * c - fi * s;                 // e^{+i theta}
    gi += fr * s + fi * c;
    t += u; if (t >= D1) t -= D1;          // t = (u*h) mod 192
  }
  G[idx * 2 + 0] = gr * (1.f / 192.f);
  G[idx * 2 + 1] = gi * (1.f / 192.f);
}

// ---- 5) irfft along axis 1 (97 -> 192) + write bf16 weights [b][co][k9][ci]
__global__ void k_irfft_w(const float* __restrict__ G, unsigned short* __restrict__ Wb) {
  __shared__ float tc[D2], tsn[D2];
  if (threadIdx.x < D2) {
    float ang = (6.2831853071795864769f / 192.f) * (float)threadIdx.x;
    float s, c;
    sincosf(ang, &s, &c);
    tc[threadIdx.x] = c;
    tsn[threadIdx.x] = s;
  }
  __syncthreads();
  int idx = blockIdx.x * 256 + threadIdx.x;
  if (idx >= BB * D1 * D2) return;
  int q = idx % D2;
  int u = (idx / D2) % D1;
  int b = idx / (D2 * D1);
  const float* g = G + (size_t)(b * D1 + u) * D2R * 2;
  float acc = g[0];                        // DC bin (imag contribution is dropped)
  int t = 0;
  for (int w = 1; w <= 95; ++w) {
    t += q; if (t >= D2) t -= D2;          // t = (q*w) mod 192
    acc += 2.f * (g[2 * w] * tc[t] - g[2 * w + 1] * tsn[t]);
  }
  acc += (q & 1) ? -g[2 * 96] : g[2 * 96]; // Nyquist bin, real part only
  acc *= (1.f / 192.f);
  // spatial[u][q] -> weight[co][ci][kh][kw]; store as [b][co][kh*3+kw][ci] bf16
  int co = u / 3, kh = u % 3, ci = q / 3, kw = q % 3;
  Wb[(((size_t)b * COUT + co) * 9 + (kh * 3 + kw)) * CIN + ci] = f2bf(acc);
}

// ---- 6) implicit-GEMM conv with WMMA bf16 ----------------------------------
//  Per workgroup: batch b, output tile 16(y) x 32(x), all 64 cout.
//  LDS: xl[612 pos][64 ci] bf16, pos row stride 144 B;  wl[64 co][9*64] bf16,
//       co stride 1168 B.  Total 162,880 B dynamic LDS.
#define XL_BYTES 88128            // 612 * 144
#define WL_STRIDE_US 584          // 1168 B / 2
__global__ __launch_bounds__(256) void k_conv(const float* __restrict__ X,
                                              const unsigned short* __restrict__ Wb,
                                              float* __restrict__ Out) {
  extern __shared__ char smem[];
  unsigned short* xl = (unsigned short*)smem;
  unsigned short* wl = (unsigned short*)(smem + XL_BYTES);
  const int tid = threadIdx.x;
  const int b  = blockIdx.z;
  const int y0 = blockIdx.y * 16;
  const int x0 = blockIdx.x * 32;

  // stage per-batch weights (74 KB, L2-resident across the 128 tiles of b)
  const unsigned short* wg = Wb + (size_t)b * (COUT * 9 * CIN);
  for (int e = tid; e < COUT * 9 * CIN; e += 256)
    wl[(e / 576) * WL_STRIDE_US + (e % 576)] = wg[e];

  // stage halo input tile, f32 -> bf16 at load time (single HBM pass over x)
  for (int ci = 0; ci < CIN; ++ci) {
    const float* xg = X + (size_t)(b * CIN + ci) * (HH * WW);
    for (int e = tid; e < 18 * 34; e += 256) {
      int ys = e / 34, xs = e % 34;
      int y = y0 + ys - 1, x = x0 + xs - 1;
      float v = 0.f;
      if ((unsigned)y < HH && (unsigned)x < WW) v = xg[y * WW + x];
      xl[e * 72 + ci] = f2bf(v);
    }
  }
  __syncthreads();

  const int wv     = tid >> 5;
  const int lane   = tid & 31;
  const int lane16 = lane & 15;
  const int half   = lane >> 4;

  // 128 C-tiles (4 mb x 32 nb); each of 8 waves computes 16 of them.
  for (int i = 0; i < 16; ++i) {
    int t  = i * 8 + wv;
    int mb = t >> 5;           // cout block (16 co)
    int nb = t & 31;           // 16 spatial positions
    int n  = nb * 16 + lane16;
    int ty = n >> 5, tx = n & 31;
    int co_a = mb * 16 + lane16;

    v8f acc = {0.f, 0.f, 0.f, 0.f, 0.f, 0.f, 0.f, 0.f};
    for (int k9 = 0; k9 < 9; ++k9) {
      int kh = k9 / 3, kw = k9 % 3;
      int pos = (ty + kh) * 34 + (tx + kw);
      // A lane layout: M = lane%16, elem e -> K = half*16 + e (16 consecutive ci)
      const unsigned short* apBase = wl + co_a * WL_STRIDE_US + k9 * 64 + half * 16;
      // B lane layout: N = lane%16, elem e -> K = half*16 + e (16 consecutive ci)
      const unsigned short* bpBase = xl + pos * 72 + half * 16;
      for (int cib = 0; cib < 2; ++cib) {  // K = 32 ci per WMMA, two blocks
        union { uint4 u[2]; v16bf v; } A, Bv;
        const uint4* ap = (const uint4*)(apBase + cib * 32);
        A.u[0]  = ap[0];   // K group base..base+7
        A.u[1]  = ap[1];   // K group base+8..base+15
        const uint4* bp = (const uint4*)(bpBase + cib * 32);
        Bv.u[0] = bp[0];
        Bv.u[1] = bp[1];
        acc = __builtin_amdgcn_wmma_f32_16x16x32_bf16(
            /*neg_a=*/false, A.v, /*neg_b=*/false, Bv.v,
            /*c_mod=*/(short)0, acc, /*reuse_a=*/false, /*reuse_b=*/false);
      }
    }
    // C layout: VGPR r, lane half -> M = half*8 + r, N = lane%16
    int y = y0 + ty, x = x0 + tx;
    float* og = Out + (((size_t)(b * COUT + mb * 16 + half * 8)) * HH + y) * WW + x;
    for (int r = 0; r < 8; ++r)
      og[(size_t)r * HH * WW] = acc[r];
  }
}

// ---------------------------------------------------------------------------
extern "C" void kernel_launch(void* const* d_in, const int* in_sizes, int n_in,
                              void* d_out, int out_size, void* d_ws, size_t ws_size,
                              hipStream_t stream) {
  (void)in_sizes; (void)n_in; (void)out_size; (void)ws_size;
  const float* x      = (const float*)d_in[0];
  const float* logits = (const float*)d_in[1];
  const float* dw     = (const float*)d_in[2];
  float* out = (float*)d_out;

  char* ws = (char*)d_ws;                       // ~3.2 MB total scratch
  double*         dist = (double*)        (ws + 0);        // 148,992 B
  int*            rank = (int*)           (ws + 149504);   //  74,496 B
  float*          fmap = (float*)         (ws + 224000);   // 1,191,936 B
  float*          G    = (float*)         (ws + 1415936);  // 1,191,936 B
  unsigned short* Wb   = (unsigned short*)(ws + 2607872);  //   589,824 B

  k_dist   <<<(NFREQ + 255) / 256, 256, 0, stream>>>(dist);
  k_rank   <<<(NFREQ + 255) / 256, 256, 0, stream>>>(dist, rank);
  k_fmap   <<<(BB * NFREQ + 255) / 256, 256, 0, stream>>>(logits, dw, rank, fmap);
  k_idft_h <<<(BB * D1 * D2R + 255) / 256, 256, 0, stream>>>(fmap, G);
  k_irfft_w<<<(BB * D1 * D2 + 255) / 256, 256, 0, stream>>>(G, Wb);

  dim3 grid(WW / 32, HH / 16, BB);              // 8 x 16 x 8 = 1024 workgroups
  k_conv<<<grid, 256, XL_BYTES + COUT * 1168, stream>>>(x, Wb, out);
}
